// Lstm_6012954215081
// MI455X (gfx1250) — matrix operand order repaired and checked by reading, not compiled, not run on hardware
//
#include <hip/hip_runtime.h>

// ---------------------------------------------------------------------------
// Problem sizes (fixed by the reference)
// ---------------------------------------------------------------------------
#define TT 512            // timesteps
#define BB 256            // batch
#define II 128            // layer-0 input width
#define HH 64             // hidden
#define GG 256            // 4*H gate width

// ---------------------------------------------------------------------------
// Types for CDNA5 WMMA (wave32): v_wmma_f32_16x16x32_bf16
// ---------------------------------------------------------------------------
typedef __bf16 bf16_t;
typedef bf16_t v16bf __attribute__((ext_vector_type(16)));
typedef float  v8f   __attribute__((ext_vector_type(8)));

union ABPack { uint4 q[2]; v16bf v; };

// ---------------------------------------------------------------------------
// Branchless fast activations (single v_exp/v_rcp/v_tanh, no EXEC games)
// ---------------------------------------------------------------------------
__device__ __forceinline__ float fast_rcp(float x) {
#if __has_builtin(__builtin_amdgcn_rcpf)
    return __builtin_amdgcn_rcpf(x);
#else
    return 1.0f / x;
#endif
}

__device__ __forceinline__ float fast_exp(float x) {
#if __has_builtin(__builtin_amdgcn_exp2f)
    return __builtin_amdgcn_exp2f(x * 1.44269504088896340736f);
#else
    return __expf(x);
#endif
}

__device__ __forceinline__ float fast_sig(float x) {
    return fast_rcp(1.0f + fast_exp(-x));
}

__device__ __forceinline__ float fast_tanh(float x) {
#if __has_builtin(__builtin_amdgcn_tanhf)
    return __builtin_amdgcn_tanhf(x);         // gfx1250 V_TANH_F32
#else
    return 2.0f * fast_sig(2.0f * x) - 1.0f;  // saturates to +-1 correctly
#endif
}

// ---------------------------------------------------------------------------
// Small utility kernels
// ---------------------------------------------------------------------------
__global__ void cvt_f32_to_bf16(const float* __restrict__ src,
                                bf16_t* __restrict__ dst, int n) {
    int i = blockIdx.x * blockDim.x + threadIdx.x;
    if (i < n) dst[i] = (bf16_t)src[i];
}

__global__ void zero_ints(int* __restrict__ p, int n) {
    int i = blockIdx.x * blockDim.x + threadIdx.x;
    if (i < n) p[i] = 0;
}

// ---------------------------------------------------------------------------
// Layer-0 input projection:  xp[t,b,n] = x[m,:].W[n,:] + bih[n] + bhh[n]
//   m = b*T + t.  One wave computes a 16(M) x 64(N) strip: 4 WMMA N-tiles
//   reusing one A operand per 32-wide K chunk.  A is f32 row-major [M,K].
// ---------------------------------------------------------------------------
__global__ void gemm_x_wmma(const float* __restrict__ A,      // [M,II] f32
                            const bf16_t* __restrict__ W,     // [GG,II] bf16
                            const float* __restrict__ bih,
                            const float* __restrict__ bhh,
                            float* __restrict__ Out)          // [T,B,GG]
{
    const int lane = threadIdx.x & 31;
    const int half = lane >> 4;
    const int l15  = lane & 15;
    const int n0   = blockIdx.x * 64;
    const int m0   = blockIdx.y * 16;
    const int am   = m0 + l15;

    v8f acc[4];
#pragma unroll
    for (int j = 0; j < 4; ++j) {
        const int ncol = n0 + j * 16 + l15;
        const float bias = bih[ncol] + bhh[ncol];
#pragma unroll
        for (int r = 0; r < 8; ++r) acc[j][r] = bias;
    }

    for (int kk = 0; kk < II; kk += 32) {
        // A operand: per-lane K chunks {klo..klo+7, klo+16..klo+23}
        const int klo = kk + half * 8;
        const float* arow = A + (size_t)am * II;
        float4 f0 = *(const float4*)(arow + klo);
        float4 f1 = *(const float4*)(arow + klo + 4);
        float4 f2 = *(const float4*)(arow + klo + 16);
        float4 f3 = *(const float4*)(arow + klo + 20);
        v16bf amat;
        amat[0]  = (bf16_t)f0.x; amat[1]  = (bf16_t)f0.y;
        amat[2]  = (bf16_t)f0.z; amat[3]  = (bf16_t)f0.w;
        amat[4]  = (bf16_t)f1.x; amat[5]  = (bf16_t)f1.y;
        amat[6]  = (bf16_t)f1.z; amat[7]  = (bf16_t)f1.w;
        amat[8]  = (bf16_t)f2.x; amat[9]  = (bf16_t)f2.y;
        amat[10] = (bf16_t)f2.z; amat[11] = (bf16_t)f2.w;
        amat[12] = (bf16_t)f3.x; amat[13] = (bf16_t)f3.y;
        amat[14] = (bf16_t)f3.z; amat[15] = (bf16_t)f3.w;

#pragma unroll
        for (int j = 0; j < 4; ++j) {
            const int ncol = n0 + j * 16 + l15;
            const bf16_t* wrow = W + (size_t)ncol * II + half * 8 + kk;
            ABPack bu;
            bu.q[0] = ((const uint4*)wrow)[0];
            bu.q[1] = ((const uint4*)wrow)[2];
            acc[j] = __builtin_amdgcn_wmma_f32_16x16x32_bf16(
                         false, amat, false, bu.v, (short)0, acc[j],
                         false, false);
        }
    }

#pragma unroll
    for (int j = 0; j < 4; ++j) {
        const int ncol = n0 + j * 16 + l15;
#pragma unroll
        for (int r = 0; r < 8; ++r) {
            const int m  = m0 + r + half * 8;
            const int bb = m >> 9, tt = m & (TT - 1);        // T = 512
            Out[((size_t)tt * BB + bb) * GG + ncol] = acc[j][r];
        }
    }
}

// ---------------------------------------------------------------------------
// Fused 3-layer pipelined LSTM scan.
// Grid: 24 blocks = 3 layers x 8 batch tiles (32 rows each), 256 threads.
// Wave (wm,wn): rows wm*16..+15, hidden slice wn*16..+15, all FOUR gate
// tiles of that slice -> cell update is pure in-register VALU, c persists
// in VGPRs for all 512 steps, h(t-1) in a double-buffered 8KB LDS buffer
// (one barrier per step).
// Layer 0 seeds gates from precomputed xp; layers 1-2 fold the input
// projection into the scan: gates = bias + h_below(t)@Wih + h_self(t-1)@Whh
// (16 WMMAs/wave/step).  Cross-layer dataflow is pipelined through global
// bf16 h-buffers guarded by per-(layer,tile) release/acquire flags, so the
// critical path is ~T+2 steps instead of 3T.
// ---------------------------------------------------------------------------
__global__ void __launch_bounds__(256)
lstm_scan(const float* __restrict__ xp,     // [T,B,GG] (layer 0 only)
          const bf16_t* __restrict__ wb,    // whh0,whh1,whh2,wih1,wih2
          const float* __restrict__ bih1, const float* __restrict__ bhh1,
          const float* __restrict__ bih2, const float* __restrict__ bhh2,
          bf16_t* __restrict__ hbufs,       // [3][T*B*HH] bf16
          int* flags)                       // [3*8], zeroed before launch
{
    __shared__ __align__(16) bf16_t sm_h[2][32][HH];

    const int tid   = threadIdx.x;
    const int lane  = tid & 31;
    const int wave  = tid >> 5;
    const int half  = lane >> 4;
    const int l15   = lane & 15;
    const int wn    = wave & 3;
    const int wm    = wave >> 2;
    const int m0    = wm * 16;
    const int hc0   = wn * 16;
    const int tile  = blockIdx.x & 7;
    const int layer = blockIdx.x >> 3;
    const int b0    = tile * 32;

    const size_t HSZ = (size_t)TT * BB * HH;
    const bf16_t* Whh = wb + (size_t)layer * GG * HH;
    const bf16_t* Wih = wb + (size_t)(3 + (layer ? layer - 1 : 0)) * GG * HH;
    const bf16_t* hin = hbufs + (size_t)(layer ? layer - 1 : 0) * HSZ;
    bf16_t* hout      = hbufs + (size_t)layer * HSZ;
    int* flag_in      = flags + (layer ? layer - 1 : 0) * 8 + tile;
    int* flag_out     = flags + layer * 8 + tile;

    // bias broadcast (layers 1,2 only; layer 0 bias folded into xp)
    float bias[4] = {0.f, 0.f, 0.f, 0.f};
    if (layer) {
        const float* bi = (layer == 1) ? bih1 : bih2;
        const float* bh = (layer == 1) ? bhh1 : bhh2;
#pragma unroll
        for (int g = 0; g < 4; ++g) {
            const int col = g * HH + hc0 + l15;
            bias[g] = bi[col] + bh[col];
        }
    }

    // Loop-invariant B operands (lane = output column, K 8h..8h+7 / 8h+16..8h+23)
    v16bf bhhm[4][2];
#pragma unroll
    for (int g = 0; g < 4; ++g) {
        const bf16_t* wrow = Whh + (size_t)(g * HH + hc0 + l15) * HH + half * 8;
#pragma unroll
        for (int kc = 0; kc < 2; ++kc) {
            ABPack u;
            const uint4* p = (const uint4*)(wrow + kc * 32);
            u.q[0] = p[0]; u.q[1] = p[2];
            bhhm[g][kc] = u.v;
        }
    }
    v16bf bihm[4][2];
    if (layer) {
#pragma unroll
        for (int g = 0; g < 4; ++g) {
            const bf16_t* wrow = Wih + (size_t)(g * HH + hc0 + l15) * HH + half * 8;
#pragma unroll
            for (int kc = 0; kc < 2; ++kc) {
                ABPack u;
                const uint4* p = (const uint4*)(wrow + kc * 32);
                u.q[0] = p[0]; u.q[1] = p[2];
                bihm[g][kc] = u.v;
            }
        }
    }

    // zero h(-1), c(-1)
    for (int e = tid; e < 32 * HH; e += 256)
        ((bf16_t*)sm_h[0])[e] = (bf16_t)0.0f;
    v8f cacc;
#pragma unroll
    for (int r = 0; r < 8; ++r) cacc[r] = 0.0f;

    __syncthreads();

    int cur = 0;
    for (int t = 0; t < TT; ++t) {
        // ---- gate accumulator init
        v8f acc[4];
        if (layer == 0) {
            const float* xr = xp + ((size_t)t * BB + b0 + m0) * GG;
#pragma unroll
            for (int g = 0; g < 4; ++g) {
                const int col = g * HH + hc0 + l15;
#pragma unroll
                for (int r = 0; r < 8; ++r)
                    acc[g][r] = xr[(size_t)(r + half * 8) * GG + col];
            }
        } else {
#pragma unroll
            for (int g = 0; g < 4; ++g)
#pragma unroll
                for (int r = 0; r < 8; ++r)
                    acc[g][r] = bias[g];
        }

        // ---- self A operand from LDS h(t-1)
        v16bf aself[2];
#pragma unroll
        for (int kc = 0; kc < 2; ++kc) {
            const int klo = kc * 32 + half * 8;
            ABPack u;
            u.q[0] = *(const uint4*)&sm_h[cur][m0 + l15][klo];
            u.q[1] = *(const uint4*)&sm_h[cur][m0 + l15][klo + 16];
            aself[kc] = u.v;
        }

        // ---- layers 1-2: consume h_below(t) from producer layer
        if (layer) {
            while (__hip_atomic_load(flag_in, __ATOMIC_ACQUIRE,
                                     __HIP_MEMORY_SCOPE_AGENT) <= t)
                __builtin_amdgcn_s_sleep(1);

            const bf16_t* hr = hin + ((size_t)t * BB + b0 + m0 + l15) * HH;
            v16bf abelow[2];
#pragma unroll
            for (int kc = 0; kc < 2; ++kc) {
                const int klo = kc * 32 + half * 8;
                ABPack u;
                u.q[0] = *(const uint4*)(hr + klo);
                u.q[1] = *(const uint4*)(hr + klo + 16);
                abelow[kc] = u.v;
            }
#pragma unroll
            for (int g = 0; g < 4; ++g) {
                acc[g] = __builtin_amdgcn_wmma_f32_16x16x32_bf16(
                             false, abelow[0], false, bihm[g][0], (short)0,
                             acc[g], false, false);
                acc[g] = __builtin_amdgcn_wmma_f32_16x16x32_bf16(
                             false, abelow[1], false, bihm[g][1], (short)0,
                             acc[g], false, false);
            }
        }

        // ---- recurrent projection
#pragma unroll
        for (int g = 0; g < 4; ++g) {
            acc[g] = __builtin_amdgcn_wmma_f32_16x16x32_bf16(
                         false, aself[0], false, bhhm[g][0], (short)0,
                         acc[g], false, false);
            acc[g] = __builtin_amdgcn_wmma_f32_16x16x32_bf16(
                         false, aself[1], false, bhhm[g][1], (short)0,
                         acc[g], false, false);
        }

        // ---- fused LSTM cell update in registers (i,f,g,o = acc[0..3])
        bf16_t hb[8];
#pragma unroll
        for (int r = 0; r < 8; ++r) {
            const float ig = fast_sig(acc[0][r]);
            const float fg = fast_sig(acc[1][r]);
            const float gg = fast_tanh(acc[2][r]);
            const float og = fast_sig(acc[3][r]);
            const float cc = fg * cacc[r] + ig * gg;
            cacc[r] = cc;
            hb[r] = (bf16_t)(og * fast_tanh(cc));
        }

        // ---- publish h(t): LDS double buffer + global h-buffer
        const int nxt = cur ^ 1;
        const int col = hc0 + l15;
#pragma unroll
        for (int r = 0; r < 8; ++r) {
            const int m = m0 + r + half * 8;
            sm_h[nxt][m][col] = hb[r];
            hout[((size_t)t * BB + (b0 + m)) * HH + col] = hb[r];
        }

        if (layer == 0 && t + 1 < TT)
            __builtin_prefetch(
                xp + ((size_t)(t + 1) * BB + b0) * GG + tid * 32, 0, 0);

        if (layer < 2) __threadfence();   // h stores visible device-wide
        __syncthreads();                  // LDS rotation + all waves done
        if (layer < 2 && tid == 0)
            __hip_atomic_store(flag_out, t + 1, __ATOMIC_RELEASE,
                               __HIP_MEMORY_SCOPE_AGENT);
        cur = nxt;
    }
}

// ---------------------------------------------------------------------------
// Final head: out[b,:] = h2[T-1,b,:] @ W_out^T + b_out   (N=2, trivial)
// ---------------------------------------------------------------------------
__global__ void final_linear(const bf16_t* __restrict__ h,   // [T,B,HH]
                             const float* __restrict__ Wout, // [2,HH]
                             const float* __restrict__ bout, // [2]
                             float* __restrict__ out)
{
    const int b = blockIdx.x * blockDim.x + threadIdx.x;
    if (b >= BB) return;
    const bf16_t* hr = h + ((size_t)(TT - 1) * BB + b) * HH;
    float s0 = bout[0], s1 = bout[1];
#pragma unroll
    for (int k = 0; k < HH; ++k) {
        const float hv = (float)hr[k];
        s0 += hv * Wout[k];
        s1 += hv * Wout[HH + k];
    }
    out[b * 2 + 0] = s0;
    out[b * 2 + 1] = s1;
}

// ---------------------------------------------------------------------------
// Launch
// ---------------------------------------------------------------------------
extern "C" void kernel_launch(void* const* d_in, const int* in_sizes, int n_in,
                              void* d_out, int out_size, void* d_ws,
                              size_t ws_size, hipStream_t stream)
{
    constexpr int M = BB * TT;

    const float* x    = (const float*)d_in[0];
    const float* Wih0 = (const float*)d_in[1];
    const float* Whh0 = (const float*)d_in[2];
    const float* bih0 = (const float*)d_in[3];
    const float* bhh0 = (const float*)d_in[4];
    const float* Wih1 = (const float*)d_in[5];
    const float* Whh1 = (const float*)d_in[6];
    const float* bih1 = (const float*)d_in[7];
    const float* bhh1 = (const float*)d_in[8];
    const float* Wih2 = (const float*)d_in[9];
    const float* Whh2 = (const float*)d_in[10];
    const float* bih2 = (const float*)d_in[11];
    const float* bhh2 = (const float*)d_in[12];
    const float* Wout = (const float*)d_in[13];
    const float* bout = (const float*)d_in[14];

    // ---- workspace layout
    char* ws = (char*)d_ws;
    const size_t OFF_HB    = (size_t)M * GG * 4;                 // after xp
    const size_t OFF_WB    = OFF_HB + 3 * (size_t)M * HH * 2;    // after hbufs
    const size_t OFF_WIH0  = OFF_WB + 5 * (size_t)GG * HH * 2;   // after scan W
    const size_t OFF_FLAGS = OFF_WIH0 + (size_t)GG * II * 2;

    float*  xp     = (float*)ws;
    bf16_t* hbufs  = (bf16_t*)(ws + OFF_HB);
    bf16_t* wb     = (bf16_t*)(ws + OFF_WB);     // whh0,whh1,whh2,wih1,wih2
    bf16_t* wb_ih0 = (bf16_t*)(ws + OFF_WIH0);
    int*    flags  = (int*)(ws + OFF_FLAGS);

    // ---- reset pipeline flags (replay-safe)
    zero_ints<<<1, 32, 0, stream>>>(flags, 24);

    // ---- weight conversion (small)
    cvt_f32_to_bf16<<<(GG * HH + 255) / 256, 256, 0, stream>>>(Whh0, wb + 0 * GG * HH, GG * HH);
    cvt_f32_to_bf16<<<(GG * HH + 255) / 256, 256, 0, stream>>>(Whh1, wb + 1 * GG * HH, GG * HH);
    cvt_f32_to_bf16<<<(GG * HH + 255) / 256, 256, 0, stream>>>(Whh2, wb + 2 * GG * HH, GG * HH);
    cvt_f32_to_bf16<<<(GG * HH + 255) / 256, 256, 0, stream>>>(Wih1, wb + 3 * GG * HH, GG * HH);
    cvt_f32_to_bf16<<<(GG * HH + 255) / 256, 256, 0, stream>>>(Wih2, wb + 4 * GG * HH, GG * HH);
    cvt_f32_to_bf16<<<(GG * II + 255) / 256, 256, 0, stream>>>(Wih0, wb_ih0, GG * II);

    // ---- layer-0 input projection (parallel WMMA GEMM)
    const dim3 gblock(32);
    const dim3 ggrid(GG / 64, M / 16);
    gemm_x_wmma<<<ggrid, gblock, 0, stream>>>(x, wb_ih0, bih0, bhh0, xp);

    // ---- fused pipelined 3-layer scan
    lstm_scan<<<24, 256, 0, stream>>>(xp, wb, bih1, bhh1, bih2, bhh2,
                                      hbufs, flags);

    // ---- head
    final_linear<<<(BB + 255) / 256, 256, 0, stream>>>(
        hbufs + 2 * (size_t)TT * BB * HH, Wout, bout, (float*)d_out);
}
